// RecurrentRetention_60413009985545
// MI455X (gfx1250) — compile-verified, matrix operand order checked by reading, not run on hardware
//
#include <hip/hip_runtime.h>
#include <hip/hip_bf16.h>
#include <cmath>

// ---------------------------------------------------------------------------
// RecurrentRetention for MI455X (gfx1250, wave32, WMMA).
// Problem: B=4, T=2048, D=1024.
//   Q = X*Wq (8192x1024x1024)  -> dominant GEMM, fused with scrambled-S scale
//   m = (sum_b vsum*x) * Wk    -> 2048x1024x1024 GEMM
//   S = gamma-scan of m (shifted), S[0]=S[1]
//   out[b,t,d] = Q[b,t,d] * S[d*2048 + t]   (reshape/transpose scramble)
// GEMMs: v_wmma_f32_16x16x32_bf16, LDS double-buffer fed by SADDR-form
// global_load_async_to_lds_b128 (ASYNCcnt), bf16 operands pre-converted,
// weights pre-transposed.  kt-loop kept rolled so accumulators stay pinned.
// ---------------------------------------------------------------------------

typedef __attribute__((ext_vector_type(16))) __bf16 v16bf;
typedef __attribute__((ext_vector_type(8)))  float  v8f;

#define GAMMA_F 0.96875f
#define DD 1024        // inner dim
#define TT 2048        // sequence
#define BB 4           // batch
#define CHUNK 128      // scan chunk length
#define NCHUNK 16      // TT / CHUNK
#define LDSS 40        // LDS row stride in ushorts (80B, conflict-free for b128)
#define BUFE (128 * LDSS)   // one LDS buffer, in ushort elements

__device__ __forceinline__ unsigned short f32_to_bf16_rne(float f) {
    unsigned int u = __float_as_uint(f);
    u += 0x7FFFu + ((u >> 16) & 1u);   // round to nearest even
    return (unsigned short)(u >> 16);
}
__device__ __forceinline__ unsigned int pack_bf16x2(float lo, float hi) {
    return (unsigned int)f32_to_bf16_rne(lo) |
           ((unsigned int)f32_to_bf16_rne(hi) << 16);
}

// CDNA5 async global->LDS copy, SADDR (GVS) form: 16B per lane, ASYNCcnt.
__device__ __forceinline__ void rr_async_cp16(unsigned ldsOff, const void* sbase,
                                              unsigned voff) {
    asm volatile("global_load_async_to_lds_b128 %0, %1, %2"
                 :: "v"(ldsOff), "v"(voff), "s"(sbase) : "memory");
}
__device__ __forceinline__ void rr_wait_async0() {
    asm volatile("s_wait_asynccnt 0x0" ::: "memory");
}

// ---------------- small reduction kernels ----------------------------------

// out[row] = sum_e W[row, e]   (one 256-thread block per row)
__global__ __launch_bounds__(256) void rr_rowsum_kernel(
    const float* __restrict__ W, float* __restrict__ out) {
    __shared__ float red[8];
    const int row = blockIdx.x;
    const float* p = W + (size_t)row * DD;
    float s = 0.f;
    for (int i = threadIdx.x; i < DD; i += 256) s += p[i];
    for (int off = 16; off > 0; off >>= 1) s += __shfl_xor(s, off, 32);
    if ((threadIdx.x & 31) == 0) red[threadIdx.x >> 5] = s;
    __syncthreads();
    if (threadIdx.x == 0) {
        float t = 0.f;
        #pragma unroll
        for (int w = 0; w < 8; ++w) t += red[w];
        out[row] = t;
    }
}

// vsum[R] = dot(x[R,:], wvs)   (one block per row R in [0, B*T))
__global__ __launch_bounds__(256) void rr_vsum_kernel(
    const float* __restrict__ x, const float* __restrict__ wvs,
    float* __restrict__ vsum) {
    __shared__ float red[8];
    const int R = blockIdx.x;
    const float* p = x + (size_t)R * DD;
    float s = 0.f;
    for (int i = threadIdx.x; i < DD; i += 256) s += p[i] * wvs[i];
    for (int off = 16; off > 0; off >>= 1) s += __shfl_xor(s, off, 32);
    if ((threadIdx.x & 31) == 0) red[threadIdx.x >> 5] = s;
    __syncthreads();
    if (threadIdx.x == 0) {
        float t = 0.f;
        #pragma unroll
        for (int w = 0; w < 8; ++w) t += red[w];
        vsum[R] = t;
    }
}

// ybf[t,i] = bf16( sum_b vsum[b,t] * x[b,t,i] )   (A-operand of the m GEMM)
__global__ __launch_bounds__(256) void rr_y_kernel(
    const float* __restrict__ x, const float* __restrict__ vsum,
    unsigned short* __restrict__ ybf) {
    const int idx = blockIdx.x * 256 + threadIdx.x;   // over T*D
    const int t = idx >> 10;
    const int i = idx & (DD - 1);
    float s = 0.f;
    #pragma unroll
    for (int b = 0; b < BB; ++b) {
        const int R = b * TT + t;
        s += vsum[R] * x[((size_t)R << 10) + i];
    }
    ybf[idx] = f32_to_bf16_rne(s);
}

// straight f32 -> bf16 conversion, 8 elements per thread
__global__ __launch_bounds__(256) void rr_cvt_bf16_kernel(
    const float* __restrict__ in, unsigned short* __restrict__ out) {
    const size_t idx = (size_t)(blockIdx.x * 256 + threadIdx.x) * 8;
    const float4 a = *(const float4*)(in + idx);
    const float4 b = *(const float4*)(in + idx + 4);
    uint4 o;
    o.x = pack_bf16x2(a.x, a.y);
    o.y = pack_bf16x2(a.z, a.w);
    o.z = pack_bf16x2(b.x, b.y);
    o.w = pack_bf16x2(b.z, b.w);
    *(uint4*)(out + idx) = o;
}

// Wt[n][k] = bf16(W[k][n])  -- 64x64 LDS-tiled convert+transpose
__global__ __launch_bounds__(256) void rr_cvtT_kernel(
    const float* __restrict__ W, unsigned short* __restrict__ Wt) {
    __shared__ __align__(16) unsigned short tile[64 * 72];  // pad 8 vs bank conflicts
    const int bi = blockIdx.x;   // k block
    const int bj = blockIdx.y;   // n block
    const int t  = threadIdx.x;
    #pragma unroll
    for (int it = 0; it < 4; ++it) {
        const int row = it * 16 + (t >> 4);      // k within tile
        const int c4  = (t & 15) * 4;            // n within tile
        const float4 v = *(const float4*)(W + (size_t)(bi * 64 + row) * DD + bj * 64 + c4);
        *(uint2*)(&tile[row * 72 + c4]) =
            make_uint2(pack_bf16x2(v.x, v.y), pack_bf16x2(v.z, v.w));
    }
    __syncthreads();
    #pragma unroll
    for (int u = 0; u < 2; ++u) {
        const int cid = t * 2 + u;
        const int n   = cid >> 3;           // n within tile
        const int kc  = (cid & 7) * 8;      // k chunk within tile
        __align__(16) unsigned short vals[8];
        #pragma unroll
        for (int j = 0; j < 8; ++j) vals[j] = tile[(kc + j) * 72 + n];
        *(uint4*)(&Wt[(size_t)(bj * 64 + n) * DD + bi * 64 + kc]) = *(const uint4*)vals;
    }
}

// ---------------- WMMA GEMM: C[M,1024] = A[M,1024] * Bt[1024,1024]^T --------
// A bf16 row-major [M][K]; Bt bf16 pre-transposed [N][K].  Block tile 128x128,
// 8 waves (4 M x 2 N), wave tile 32x64, K-step 32.  Double-buffered LDS fed
// by async copies; kt loop NOT unrolled so acc regs stay pinned (no phi moves).
// FUSE epilogue: C[row,col] *= Sbuf[col*2048 + (row & 2047)].
template <bool FUSE>
__global__ __launch_bounds__(256) void rr_gemm_bf16_kernel(
    const unsigned short* __restrict__ A, const unsigned short* __restrict__ Bt,
    float* __restrict__ C, const float* __restrict__ Sbuf) {
    __shared__ __align__(16) unsigned short ldsA[2 * BUFE];
    __shared__ __align__(16) unsigned short ldsB[2 * BUFE];

    const int tid   = threadIdx.x;
    const int lane  = tid & 31;
    const int wave  = tid >> 5;
    const int waveM = wave & 3;      // 0..3
    const int waveN = wave >> 2;     // 0..1
    const int l15   = lane & 15;
    const int half  = lane >> 4;
    const int blockM = blockIdx.x * 128;
    const int blockN = blockIdx.y * 128;

    const unsigned aBase = (unsigned)(uintptr_t)(&ldsA[0]);
    const unsigned bBase = (unsigned)(uintptr_t)(&ldsB[0]);

    // per-thread copy slots: 2x16B of A and 2x16B of B per k-step
    // chunk id -> row = cid>>2 (0..127), kc = (cid&3)*8 bf16 elems
    unsigned ldsOffA[2], ldsOffB[2], gOffA[2], gOffB[2];
    #pragma unroll
    for (int u = 0; u < 2; ++u) {
        const int cid = tid * 2 + u;
        const int row = cid >> 2;
        const int kc  = (cid & 3) * 8;
        ldsOffA[u] = aBase + 2u * (unsigned)(row * LDSS + kc);
        ldsOffB[u] = bBase + 2u * (unsigned)(row * LDSS + kc);
        gOffA[u] = 2u * (unsigned)((blockM + row) * DD + kc);   // byte offsets
        gOffB[u] = 2u * (unsigned)((blockN + row) * DD + kc);
    }

    v8f acc[2][4];
    #pragma unroll
    for (int mi = 0; mi < 2; ++mi)
        #pragma unroll
        for (int ni = 0; ni < 4; ++ni)
            acc[mi][ni] = {};

    union Frag { uint4 u[2]; v16bf v; };

    auto issueCopies = [&](unsigned kByte, unsigned bufByte) {
        #pragma unroll
        for (int u = 0; u < 2; ++u) {
            rr_async_cp16(ldsOffA[u] + bufByte, A, gOffA[u] + kByte);
            rr_async_cp16(ldsOffB[u] + bufByte, Bt, gOffB[u] + kByte);
        }
    };
    auto compute = [&](unsigned curE) {   // curE: ushort offset of live buffer
        Frag fa[2], fb[4];
        #pragma unroll
        for (int mi = 0; mi < 2; ++mi) {
            const unsigned short* p = &ldsA[curE + (waveM * 32 + mi * 16 + l15) * LDSS];
            fa[mi].u[0] = *(const uint4*)(p + 8 * half);        // k = 8h .. 8h+7
            fa[mi].u[1] = *(const uint4*)(p + 16 + 8 * half);   // k = 16+8h .. +7
        }
        #pragma unroll
        for (int ni = 0; ni < 4; ++ni) {
            const unsigned short* p = &ldsB[curE + (waveN * 64 + ni * 16 + l15) * LDSS];
            fb[ni].u[0] = *(const uint4*)(p + 8 * half);
            fb[ni].u[1] = *(const uint4*)(p + 16 + 8 * half);
        }
        #pragma unroll
        for (int mi = 0; mi < 2; ++mi)
            #pragma unroll
            for (int ni = 0; ni < 4; ++ni)
                acc[mi][ni] = __builtin_amdgcn_wmma_f32_16x16x32_bf16(
                    false, fa[mi].v, false, fb[ni].v,
                    (short)0, acc[mi][ni], false, false);
    };

    // prologue: stage k-step 0 into buffer 0
    issueCopies(0u, 0u);
    rr_wait_async0();
    __syncthreads();

    unsigned curE = 0;                       // ushort offset of live buffer
    #pragma clang loop unroll(disable)
    for (int kt = 0; kt < DD / 32 - 1; ++kt) {
        issueCopies((unsigned)(kt + 1) * 64u, (curE ^ BUFE) * 2u);  // next tile
        compute(curE);
        rr_wait_async0();     // our copies into the other buffer are done
        __syncthreads();      // everyone's are visible
        curE ^= BUFE;
    }
    compute(curE);            // tail k-step, no staging

    // ---- epilogue: C layout is VGPR v -> M = v + 8*half, N = lane&15 ----
    const int mbase = blockM + waveM * 32;
    const int nbase = blockN + waveN * 64;
    #pragma unroll
    for (int mi = 0; mi < 2; ++mi) {
        #pragma unroll
        for (int ni = 0; ni < 4; ++ni) {
            const int col = nbase + ni * 16 + l15;
            const int rbase = mbase + mi * 16 + 8 * half;   // rows rbase..rbase+7
            if (FUSE) {
                // 8 consecutive t at fixed col -> two float4 loads
                const int tbase = rbase & (TT - 1);
                const float* sp = Sbuf + ((size_t)col << 11) + tbase;
                const float4 s0 = *(const float4*)(sp);
                const float4 s1 = *(const float4*)(sp + 4);
                acc[mi][ni][0] *= s0.x; acc[mi][ni][1] *= s0.y;
                acc[mi][ni][2] *= s0.z; acc[mi][ni][3] *= s0.w;
                acc[mi][ni][4] *= s1.x; acc[mi][ni][5] *= s1.y;
                acc[mi][ni][6] *= s1.z; acc[mi][ni][7] *= s1.w;
            }
            #pragma unroll
            for (int v = 0; v < 8; ++v)
                C[(size_t)(rbase + v) * DD + col] = acc[mi][ni][v];
        }
    }
}

// ---------------- gamma scan (chunked), in place on m/S buffer --------------

// local inclusive scan within each 128-chunk; m[0] treated as 0 (shift quirk)
__global__ __launch_bounds__(256) void rr_scan_local_kernel(float* __restrict__ S) {
    const int idx = blockIdx.x * 256 + threadIdx.x;  // over NCHUNK * D
    const int d = idx & (DD - 1);
    const int c = idx >> 10;
    float acc = 0.f;
    for (int j = 0; j < CHUNK; ++j) {
        const int t = c * CHUNK + j;
        const size_t o = ((size_t)t << 10) + d;
        const float v = (t == 0) ? 0.f : S[o];
        acc = GAMMA_F * acc + v;
        S[o] = acc;
    }
}

// serial scan of the 16 chunk carries per column d
__global__ __launch_bounds__(256) void rr_carry_scan_kernel(
    const float* __restrict__ S, float* __restrict__ Cp, float gammaL) {
    const int d = blockIdx.x * 256 + threadIdx.x;    // 0..1023
    float E = 0.f;
    for (int c = 0; c < NCHUNK; ++c) {
        const float endv = S[((size_t)((c + 1) * CHUNK - 1) << 10) + d];
        E = endv + gammaL * E;
        Cp[(size_t)c * DD + d] = E;
    }
}

// apply carried state: S[t] += gamma^(j+1) * E[chunk-1]
__global__ __launch_bounds__(256) void rr_carry_apply_kernel(
    float* __restrict__ S, const float* __restrict__ Cp) {
    const int idx = blockIdx.x * 256 + threadIdx.x;  // over T*D
    const int t = idx >> 10;
    const int d = idx & (DD - 1);
    const int c = t >> 7;
    if (c > 0) {
        const int j = t & (CHUNK - 1);
        S[idx] += __powf(GAMMA_F, (float)(j + 1)) * Cp[(size_t)(c - 1) * DD + d];
    }
}

// S[0,:] = S[1,:]  (reference prepends states[:1])
__global__ __launch_bounds__(256) void rr_fix_row0_kernel(float* __restrict__ S) {
    const int d = blockIdx.x * 256 + threadIdx.x;
    if (d < DD) S[d] = S[DD + d];
}

// ---------------------------------------------------------------------------

extern "C" void kernel_launch(void* const* d_in, const int* in_sizes, int n_in,
                              void* d_out, int out_size, void* d_ws, size_t ws_size,
                              hipStream_t stream) {
    const float* x  = (const float*)d_in[0];   // [4,2048,1024]
    const float* Wq = (const float*)d_in[1];   // [1024,1024]
    const float* Wk = (const float*)d_in[2];
    const float* Wv = (const float*)d_in[3];
    float* out = (float*)d_out;                // [4,2048,1024]

    // workspace layout (~32.2 MB)
    float* S   = (float*)d_ws;                         // [2048,1024] m then S
    float* Cp  = S   + (size_t)TT * DD;                // [16,1024]
    float* wvs = Cp  + (size_t)NCHUNK * DD;            // [1024]
    float* vs  = wvs + DD;                             // [8192]
    unsigned short* xbf = (unsigned short*)(vs + BB * TT);   // [8192,1024] bf16
    unsigned short* wqt = xbf + (size_t)BB * TT * DD;        // [1024,1024] bf16 (transposed)
    unsigned short* wkt = wqt + (size_t)DD * DD;             // [1024,1024] bf16 (transposed)
    unsigned short* ybf = wkt + (size_t)DD * DD;             // [2048,1024] bf16

    const float gammaL = powf(GAMMA_F, (float)CHUNK);

    // 1) wvs[i] = sum_e Wv[i,e];  vs[R] = x[R,:] . wvs
    rr_rowsum_kernel<<<DD, 256, 0, stream>>>(Wv, wvs);
    rr_vsum_kernel<<<BB * TT, 256, 0, stream>>>(x, wvs, vs);
    // 2) precision/layout prep: x -> bf16; Wq,Wk -> bf16 transposed
    rr_cvt_bf16_kernel<<<(BB * TT * DD) / (256 * 8), 256, 0, stream>>>(x, xbf);
    rr_cvtT_kernel<<<dim3(DD / 64, DD / 64), 256, 0, stream>>>(Wq, wqt);
    rr_cvtT_kernel<<<dim3(DD / 64, DD / 64), 256, 0, stream>>>(Wk, wkt);
    // 3) ybf[t,i] = bf16(sum_b vs[b,t] * x[b,t,i])
    rr_y_kernel<<<(TT * DD) / 256, 256, 0, stream>>>(x, vs, ybf);
    // 4) m = y @ Wk  (WMMA, plain store into S buffer)
    rr_gemm_bf16_kernel<false><<<dim3(TT / 128, DD / 128), 256, 0, stream>>>(ybf, wkt, S, nullptr);
    // 5) gamma scan with shift + row0 fixup
    rr_scan_local_kernel<<<(NCHUNK * DD) / 256, 256, 0, stream>>>(S);
    rr_carry_scan_kernel<<<DD / 256, 256, 0, stream>>>(S, Cp, gammaL);
    rr_carry_apply_kernel<<<(TT * DD) / 256, 256, 0, stream>>>(S, Cp);
    rr_fix_row0_kernel<<<DD / 256, 256, 0, stream>>>(S);
    // 6) out = (X @ Wq) * S_scrambled   (WMMA, fused epilogue)
    rr_gemm_bf16_kernel<true><<<dim3((BB * TT) / 128, DD / 128), 256, 0, stream>>>(xbf, wqt, out, S);
}
